// BAE_k_deform_k_bae_corse2fine_affine_44263932952974
// MI455X (gfx1250) — matrix operand order repaired and checked by reading, not compiled, not run on hardware
//
#include <hip/hip_runtime.h>
#include <hip/hip_bf16.h>
#include <math.h>

// ---------------------------------------------------------------------------
// Model constants (from reference): T=4 parts, PD=64, ZD=256, GF=256
// bs=4, nq=8192 -> ROWS=32768 decoder rows. hid = T*PD/2 = 128.
// ---------------------------------------------------------------------------
#define T_PARTS 4
#define PD      64
#define ZD      256
#define GF      256
#define BS      4
#define NQ      8192
#define ROWS    (BS * NQ)          // 32768
#define HID     128
#define BNS     0.9999950000374997f   // 1/sqrt(1+1e-5)

// ---------------------------- WMMA GEMM ------------------------------------
#define BM 128
#define BN 64
#define BK 32

typedef __attribute__((ext_vector_type(16))) __bf16 v16bf;
typedef __attribute__((ext_vector_type(8)))  float  v8f;
typedef __attribute__((ext_vector_type(4)))  unsigned int v4u;
typedef __attribute__((ext_vector_type(8)))  int v8i;
typedef __attribute__((ext_vector_type(4)))  int v4i;

union Frag { unsigned int u[8]; v16bf v; };
union Acc  { float f[8];        v8f  v; };

__device__ __forceinline__ unsigned short f2bf(float f) {
  unsigned int u = __float_as_uint(f);
  u += 0x7FFFu + ((u >> 16) & 1u);     // round-to-nearest-even
  return (unsigned short)(u >> 16);
}
// fast pack: round-half-up on both halves, one v_perm_b32 to gather hi16s
__device__ __forceinline__ unsigned int pk2_rn(float a, float b) {
  unsigned int ua = __float_as_uint(a) + 0x8000u;
  unsigned int ub = __float_as_uint(b) + 0x8000u;
  // bytes 4..7 = ub, bytes 0..3 = ua ; sel picks {ub.hi16, ua.hi16}
  return __builtin_amdgcn_perm(ub, ua, 0x07060302u);
}

// generic pointer (to a __shared__ object) -> 32-bit LDS offset for async/TDM
__device__ __forceinline__ unsigned lds_off32(const void* p) {
  return (unsigned)(size_t)(__attribute__((address_space(3))) const void*)p;
}

__device__ __forceinline__ float act_f(float v, int act) {
  switch (act) {
    case 1: return v > 0.f ? v : 0.01f * v;          // leaky relu
    case 2: return 1.f / (1.f + __expf(-v));          // sigmoid
    case 3: return tanhf(v);                          // tanh
    case 4: return fmaxf(v, 0.f);                     // relu
  }
  return v;
}

// C[M,N](ldc) = act(A[M,K](lda) @ W[K,N] + bias)
// A: global_load_async_to_lds (f32) -> cooperative bf16 pack (ASYNCcnt path)
// W: TENSOR_LOAD_TO_LDS 2D descriptor, HW LDS padding (TENSORcnt path)
// compute: v_wmma_f32_16x16x32_bf16, f32 accumulate, fused bias+activation.
__global__ __launch_bounds__(256)
void k_wmma_gemm(const float* __restrict__ A, int lda,
                 const float* __restrict__ W,
                 const float* __restrict__ bias,
                 float* __restrict__ C, int ldc,
                 int M, int N, int K, int act) {
  __shared__ float          sAf[BM][BK + 1];     // f32 A tile (async dst)
  __shared__ unsigned short sA[BM][BK + 2];      // bf16 A, row-major, padded
  __shared__ float          sBf[BK][BN + 1];     // f32 W tile (TDM dst, HW pad)
  __shared__ unsigned int   sBp[BK / 2][BN + 1]; // packed bf16 K-pairs

  const int tid     = threadIdx.x;
  const int rowBase = blockIdx.y * BM;
  const int colBase = blockIdx.x * BN;
  const int wid  = tid >> 5;
  const int lane = tid & 31;
  const int wm   = (wid >> 1) * 32;   // wave's M offset in tile (4 rows of waves)
  const int wn   = (wid & 1) * 32;    // wave's N offset in tile (2 cols of waves)
  const int half = lane >> 4;         // 0: lanes 0-15, 1: lanes 16-31
  const int lr   = lane & 15;
  const bool rowFull = (rowBase + BM <= M);
  const bool colFull = (colBase + BN <= N);

  Acc acc[2][2];
#pragma unroll
  for (int mt = 0; mt < 2; ++mt)
#pragma unroll
    for (int nt = 0; nt < 2; ++nt)
#pragma unroll
      for (int r = 0; r < 8; ++r) acc[mt][nt].f[r] = 0.f;

  for (int k0 = 0; k0 < K; k0 += BK) {
    const bool kFull = (k0 + BK <= K);
    const bool fastA = kFull && rowFull;
    const bool fastB = kFull && colFull;

    // ---- stage A tile (128x32 f32) via async global->LDS DMA ----
    if (fastA) {
#pragma unroll
      for (int i = 0; i < 16; ++i) {
        int e = tid + i * 256;
        int m = e >> 5, k = e & 31;
        unsigned long long ga = (unsigned long long)(const void*)
            (A + (size_t)(rowBase + m) * lda + (k0 + k));
        unsigned lo = lds_off32(&sAf[m][k]);
        asm volatile("global_load_async_to_lds_b32 %0, %1, off"
                     :: "v"(lo), "v"(ga) : "memory");
      }
    } else {
      // edge tile: clamp address (always in-bounds), select 0; write bf16 direct
#pragma unroll
      for (int i = 0; i < 16; ++i) {
        int e  = tid + i * 256;
        int m  = e >> 5, k = e & 31;
        int gm = rowBase + m, gk = k0 + k;
        int gmc = gm < M ? gm : M - 1;
        int gkc = gk < K ? gk : K - 1;
        float v = A[(size_t)gmc * lda + gkc];
        sA[m][k] = (gm < M && gk < K) ? f2bf(v) : (unsigned short)0;
      }
    }

    // ---- stage B tile (32x64 f32) via Tensor Data Mover ----
    if (fastB) {
      if (wid == 0) {
        unsigned long long ga = (unsigned long long)(const void*)
            (W + (size_t)k0 * N + colBase);        // tile start, byte address
        // D# group0: count=1 | lds_addr | global_addr | type=2
        v4u g0;
        g0[0] = 1u;                                 // count=1 (valid), no gather
        g0[1] = lds_off32(&sBf[0][0]);              // LDS dest
        g0[2] = (unsigned)(ga & 0xFFFFFFFFull);
        g0[3] = (unsigned)((ga >> 32) & 0x01FFFFFFull) | (2u << 30);  // type=2
        // D# group1: data_size=4B(code 2), pad: 1 DWORD every 64 DWORDs
        v8i g1;
        g1[0] = (int)((2u << 16) | (1u << 20) | (5u << 22));  // ds|pad_en|intvl
        g1[1] = (int)((unsigned)(N & 0xFFFF) << 16);          // tensor_dim0 lo
        g1[2] = (int)(((unsigned)N >> 16) | ((unsigned)(K & 0xFFFF) << 16));
        g1[3] = (int)(((unsigned)K >> 16) | (64u << 16));     // tile_dim0=64
        g1[4] = 32;                                           // tile_dim1=32
        g1[5] = N;                                            // dim0_stride lo
        g1[6] = 0;                                            // stride hi / dim1_stride
        g1[7] = 0;
        v4i gz4 = {0, 0, 0, 0};                               // 2D: groups 2/3 unused
        v8i gz8 = {0, 0, 0, 0, 0, 0, 0, 0};
        __builtin_amdgcn_tensor_load_to_lds(g0, g1, gz4, gz4, gz8, 0);
      }
    } else {
#pragma unroll
      for (int j = 0; j < 8; ++j) {
        int idx = tid + j * 256;
        int k = idx >> 6, n = idx & 63;
        int gk = k0 + k, gn = colBase + n;
        int gkc = gk < K ? gk : K - 1;
        int gnc = gn < N ? gn : N - 1;
        float v = W[(size_t)gkc * N + gnc];
        sBf[k][n] = (gk < K && gn < N) ? v : 0.f;
      }
    }

    // prefetch next A K-tile into caches (global_prefetch_b8)
    if (k0 + BK < K) {
      int gm = rowBase + (tid >> 1);
      int gk = k0 + BK + (tid & 1) * 16;
      if (gm < M && gk < K) __builtin_prefetch(&A[(size_t)gm * lda + gk], 0, 0);
    }

    asm volatile("s_wait_asynccnt 0" ::: "memory");  // this wave's async loads
    __builtin_amdgcn_s_wait_tensorcnt(0);            // wave 0's TDM op
    __syncthreads();

    // ---- cooperative packs: f32 -> packed bf16 pairs ----
    if (fastA) {
#pragma unroll
      for (int j = 0; j < 8; ++j) {
        int idx = tid + j * 256;            // 2048 pairs
        int m = idx >> 4, kp = idx & 15;
        *(unsigned int*)&sA[m][2 * kp] = pk2_rn(sAf[m][2 * kp], sAf[m][2 * kp + 1]);
      }
    }
#pragma unroll
    for (int j = 0; j < 4; ++j) {
      int idx = tid + j * 256;              // 1024 pairs
      int kp = idx >> 6, n = idx & 63;
      sBp[kp][n] = pk2_rn(sBf[2 * kp][n], sBf[2 * kp + 1][n]);
    }
    __syncthreads();

    // ---- build fragments per documented CDNA5 VGPR layouts ----
    Frag fa[2], fb[2];
#pragma unroll
    for (int mt = 0; mt < 2; ++mt) {
      const unsigned short* rp = &sA[wm + mt * 16 + lr][0];
#pragma unroll
      for (int v = 0; v < 4; ++v) {
        // 16-bit A 16x32: VGPR v(0..3): K = 8*half + 2v,2v+1 ; v(4..7): +16
        fa[mt].u[v]     = *(const unsigned int*)(rp + half * 8 + 2 * v);
        fa[mt].u[4 + v] = *(const unsigned int*)(rp + 16 + half * 8 + 2 * v);
      }
    }
#pragma unroll
    for (int nt = 0; nt < 2; ++nt) {
      int n = wn + nt * 16 + lr;
#pragma unroll
      for (int v = 0; v < 8; ++v)   // B 32x16: VGPR v(0..3): K = 8*half + 2v,2v+1 ; v(4..7): +16
        fb[nt].u[v] = sBp[(v & 3) + half * 4 + (v >> 2) * 8][n];
    }

#pragma unroll
    for (int mt = 0; mt < 2; ++mt)
#pragma unroll
      for (int nt = 0; nt < 2; ++nt)
        acc[mt][nt].v = __builtin_amdgcn_wmma_f32_16x16x32_bf16(
            false, fa[mt].v, false, fb[nt].v, (short)0, acc[mt][nt].v,
            false, false);
    __syncthreads();
  }

  // ---- fused epilogue: bias + activation, store f32 ----
#pragma unroll
  for (int mt = 0; mt < 2; ++mt)
#pragma unroll
    for (int nt = 0; nt < 2; ++nt) {
      int col = colBase + wn + nt * 16 + lr;
#pragma unroll
      for (int r = 0; r < 8; ++r) {
        int row = rowBase + wm + mt * 16 + half * 8 + r;  // C: VGPR r -> M = r + 8*half
        if (row < M && col < N) {
          float v = acc[mt][nt].f[r];
          if (bias) v += bias[col];
          C[(size_t)row * ldc + col] = act_f(v, act);
        }
      }
    }
}

// ------------------------- encoder: FPS ------------------------------------
__global__ void k_fps(const float* __restrict__ xyz, int N, int npoint,
                      int* __restrict__ fidx) {
  int b = blockIdx.x;
  const float* p = xyz + (size_t)b * N * 3;
  __shared__ float dist[2048];
  __shared__ float rmax[256];
  __shared__ int   rarg[256];
  __shared__ int   s_far;
  int tid = threadIdx.x;
  for (int j = tid; j < N; j += 256) dist[j] = 1e10f;
  if (tid == 0) s_far = 0;
  __syncthreads();
  for (int it = 0; it < npoint; ++it) {
    int far = s_far;
    if (tid == 0) fidx[b * npoint + it] = far;
    float cx = p[far * 3], cy = p[far * 3 + 1], cz = p[far * 3 + 2];
    float bm = -1.f; int ba = 0;
    for (int j = tid; j < N; j += 256) {
      float dx = p[j * 3] - cx, dy = p[j * 3 + 1] - cy, dz = p[j * 3 + 2] - cz;
      float dd = fminf(dist[j], dx * dx + dy * dy + dz * dz);
      dist[j] = dd;
      if (dd > bm) { bm = dd; ba = j; }
    }
    rmax[tid] = bm; rarg[tid] = ba;
    __syncthreads();
    for (int st = 128; st > 0; st >>= 1) {
      if (tid < st) {
        if (rmax[tid + st] > rmax[tid] ||
            (rmax[tid + st] == rmax[tid] && rarg[tid + st] < rarg[tid])) {
          rmax[tid] = rmax[tid + st]; rarg[tid] = rarg[tid + st];
        }
      }
      __syncthreads();
    }
    if (tid == 0) s_far = rarg[0];
    __syncthreads();
  }
}

// ------------- encoder: ball query + group + 3-layer MLP + max -------------
__global__ void k_sa(const float* __restrict__ xyz, const float* __restrict__ pts,
                     int N, int Cpts, const int* __restrict__ fidx, int S,
                     float radius,
                     const float* __restrict__ w1, const float* __restrict__ b1, int C1,
                     const float* __restrict__ w2, const float* __restrict__ b2, int C2,
                     const float* __restrict__ w3, const float* __restrict__ b3, int C3,
                     float* __restrict__ newxyz, float* __restrict__ outF) {
  const int KNB = 32;
  int bs = blockIdx.x / S;
  int s  = blockIdx.x % S;
  int tid = threadIdx.x;
  int wave = tid >> 5, lane = tid & 31;
  __shared__ int nbr[KNB];
  __shared__ unsigned char flag[2048];
  __shared__ float bufA[8][288];
  __shared__ float bufB[8][288];
  __shared__ float wmx[8][256];
  const float* X = xyz + (size_t)bs * N * 3;
  const float* P = pts + (size_t)bs * N * Cpts;
  int ctr = fidx[bs * S + s];
  float cx = X[ctr * 3], cy = X[ctr * 3 + 1], cz = X[ctr * 3 + 2];
  if (tid < 3) newxyz[(size_t)(bs * S + s) * 3 + tid] = X[ctr * 3 + tid];
  float r2 = radius * radius;
  for (int j = tid; j < N; j += 256) {
    float dx = X[j * 3] - cx, dy = X[j * 3 + 1] - cy, dz = X[j * 3 + 2] - cz;
    flag[j] = (dx * dx + dy * dy + dz * dz <= r2) ? 1 : 0;
  }
  for (int j = tid; j < 8 * 256; j += 256) wmx[j >> 8][j & 255] = -1e30f;
  __syncthreads();
  if (tid == 0) {   // first-KNB indices inside the ball (matches sorted-index take)
    int c = 0;
    for (int j = 0; j < N && c < KNB; ++j) if (flag[j]) nbr[c++] = j;
    int f = (c > 0) ? nbr[0] : ctr;
    for (; c < KNB; ++c) nbr[c] = f;
  }
  __syncthreads();
  int Cin0 = 3 + Cpts;
  for (int kk = 0; kk < 4; ++kk) {   // each wave handles 4 of the 32 neighbors
    int n = nbr[wave + kk * 8];
    for (int c = lane; c < Cin0; c += 32) {
      float v;
      if (c < 3) v = X[n * 3 + c] - ((c == 0) ? cx : (c == 1) ? cy : cz);
      else       v = P[(size_t)n * Cpts + (c - 3)];
      bufA[wave][c] = v;
    }
    __syncthreads();
    for (int co = lane; co < C1; co += 32) {
      float acc = b1[co];
      for (int ci = 0; ci < Cin0; ++ci) acc += bufA[wave][ci] * w1[ci * C1 + co];
      bufB[wave][co] = fmaxf(acc * BNS, 0.f);
    }
    __syncthreads();
    for (int co = lane; co < C2; co += 32) {
      float acc = b2[co];
      for (int ci = 0; ci < C1; ++ci) acc += bufB[wave][ci] * w2[ci * C2 + co];
      bufA[wave][co] = fmaxf(acc * BNS, 0.f);
    }
    __syncthreads();
    for (int co = lane; co < C3; co += 32) {
      float acc = b3[co];
      for (int ci = 0; ci < C2; ++ci) acc += bufA[wave][ci] * w3[ci * C3 + co];
      wmx[wave][co] = fmaxf(wmx[wave][co], fmaxf(acc * BNS, 0.f));
    }
    __syncthreads();
  }
  for (int co = tid; co < C3; co += 256) {
    float m = wmx[0][co];
    for (int w = 1; w < 8; ++w) m = fmaxf(m, wmx[w][co]);
    outF[(size_t)(bs * S + s) * C3 + co] = m;
  }
}

// --------------------------- small helpers ---------------------------------
__global__ void k_z(const float* __restrict__ pts, const float* __restrict__ w,
                    const float* __restrict__ b0, float* __restrict__ z) {
  int i = blockIdx.x * 256 + threadIdx.x;   // BS*ZD = 1024
  if (i >= BS * ZD) return;
  int bb = i / ZD, c = i % ZD;
  float acc = b0[0];
  for (int n = 0; n < 16; ++n) acc += pts[(size_t)(bb * 16 + n) * ZD + c] * w[n];
  z[i] = acc;
}

__global__ void k_dense_small(const float* __restrict__ X, const float* __restrict__ W,
                              const float* __restrict__ B, float* __restrict__ Y,
                              int R, int Ci, int Co, int act) {
  int total = R * Co;
  for (int e = threadIdx.x; e < total; e += 256) {
    int r = e / Co, c = e % Co;
    float acc = B ? B[c] : 0.f;
    for (int i = 0; i < Ci; ++i) acc += X[r * Ci + i] * W[i * Co + c];
    Y[e] = act_f(acc, act);
  }
}

__global__ void k_add_vec(const float* __restrict__ a, int astride,
                          const float* __restrict__ b, float* __restrict__ y,
                          int rows, int cols) {
  int i = blockIdx.x * 256 + threadIdx.x;
  if (i >= rows * cols) return;
  int r = i / cols, c = i % cols;
  y[i] = a[r * astride + c] + b[c];
}

// inp[row] = [point_coord(3), ezd[b]+part_code[t] (64)]
__global__ void k_build_dinp(const float* __restrict__ pc, const float* __restrict__ ezd,
                             const float* __restrict__ pcodes, float* __restrict__ out,
                             int t) {
  long long i = (long long)blockIdx.x * 256 + threadIdx.x;
  if (i >= (long long)ROWS * 67) return;
  int r = (int)(i / 67), c = (int)(i % 67);
  int b = r / NQ;
  out[i] = (c < 3) ? pc[(size_t)r * 3 + c]
                   : ezd[b * ZD + t * PD + (c - 3)] + pcodes[t * PD + (c - 3)];
}

// inp[row] = [pc + clip(tanh(def)*sigmoid(pp[3:])+tanh(pp[:3])), zf[b] (64)]
__global__ void k_build_binp(const float* __restrict__ pc, const float* __restrict__ def3,
                             const float* __restrict__ pp6, const float* __restrict__ zf,
                             float* __restrict__ out) {
  long long i = (long long)blockIdx.x * 256 + threadIdx.x;
  if (i >= (long long)ROWS * 67) return;
  int r = (int)(i / 67), c = (int)(i % 67);
  int b = r / NQ;
  float v;
  if (c < 3) {
    float tr = tanhf(pp6[c]);
    float sc = 1.f / (1.f + __expf(-pp6[3 + c]));
    float d  = def3[(size_t)r * 3 + c] * sc + tr;   // def3 already tanh'ed by GEMM
    d = fminf(1.f, fmaxf(-1.f, d));
    v = pc[(size_t)r * 3 + c] + d;
  } else {
    v = zf[b * PD + (c - 3)];
  }
  out[i] = v;
}

// fine input: [pc(3), pfe(1024)]
__global__ void k_build_fine(const float* __restrict__ pc, const float* __restrict__ pfe,
                             float* __restrict__ out) {
  long long i = (long long)blockIdx.x * 256 + threadIdx.x;
  if (i >= (long long)ROWS * 1027) return;
  int r = (int)(i / 1027), c = (int)(i % 1027);
  out[i] = (c < 3) ? pc[(size_t)r * 3 + c] : pfe[(size_t)r * 1024 + (c - 3)];
}

__global__ void k_maxgs(const float* __restrict__ gs, float* __restrict__ out) {
  int r = blockIdx.x * 256 + threadIdx.x;
  if (r >= ROWS) return;
  float m = gs[r * 4];
  for (int t = 1; t < 4; ++t) m = fmaxf(m, gs[r * 4 + t]);
  out[r] = m;
}

// ---------------------------------------------------------------------------
static inline int cdiv(long long a, long long b) { return (int)((a + b - 1) / b); }

static void gemm(hipStream_t s, const float* A, int lda, const float* W,
                 const float* bias, float* C, int ldc, int M, int N, int K, int act) {
  dim3 g(cdiv(N, BN), cdiv(M, BM));
  k_wmma_gemm<<<g, dim3(256), 0, s>>>(A, lda, W, bias, C, ldc, M, N, K, act);
}

extern "C" void kernel_launch(void* const* d_in, const int* in_sizes, int n_in,
                              void* d_out, int out_size, void* d_ws, size_t ws_size,
                              hipStream_t stream) {
  // ---- input order: jax pytree flatten, sorted dict keys ----
  const float* fps_point = (const float*)d_in[0];
  const float* d_b1 = (const float*)d_in[1];   // (T,256)
  const float* d_b2 = (const float*)d_in[2];   // (T,256)
  const float* d_b3 = (const float*)d_in[3];   // (T,3)
  const float* de1w0 = (const float*)d_in[4];  // (256,128)
  const float* de1b0 = (const float*)d_in[5];
  const float* de1w1 = (const float*)d_in[6];  // (128,128)
  const float* de1b1 = (const float*)d_in[7];
  const float* de1w2 = (const float*)d_in[8];  // (128,256)
  const float* de1b2 = (const float*)d_in[9];
  const float* pcodes = (const float*)d_in[10];  // (T,64)
  const float* d_w1 = (const float*)d_in[11];  // (T,67,256)
  const float* d_w2 = (const float*)d_in[12];  // (T,256,256)
  const float* d_w3 = (const float*)d_in[13];  // (T,256,3)
  const float* conv0_b = (const float*)d_in[14];
  const float* conv0_w = (const float*)d_in[15];
  const float* saw[4][3], *sab[4][3];
  for (int l = 0; l < 4; ++l)
    for (int j = 0; j < 3; ++j) {
      saw[l][j] = (const float*)d_in[16 + l * 6 + j * 2];
      sab[l][j] = (const float*)d_in[16 + l * 6 + j * 2 + 1];
    }
  const float* g_bb1 = (const float*)d_in[40];  // (T,1024)
  const float* g_bb2 = (const float*)d_in[41];  // (T,256)
  const float* g_bb3 = (const float*)d_in[42];  // (T,1)
  const float* g_bw1 = (const float*)d_in[43];  // (T,67,1024)
  const float* g_bw2 = (const float*)d_in[44];  // (T,1024,256)
  const float* g_bw3 = (const float*)d_in[45];  // (T,256,1)
  const float* ge1w0 = (const float*)d_in[46];
  const float* ge1b0 = (const float*)d_in[47];
  const float* ge1w1 = (const float*)d_in[48];
  const float* ge1b1 = (const float*)d_in[49];
  const float* ge1w2 = (const float*)d_in[50];
  const float* ge1b2 = (const float*)d_in[51];
  const float* emb2b = (const float*)d_in[52];
  const float* emb2w = (const float*)d_in[53];
  const float* emb3b = (const float*)d_in[54];
  const float* emb3w = (const float*)d_in[55];
  const float* fine_b1 = (const float*)d_in[56];
  const float* fine_b2 = (const float*)d_in[57];
  const float* fine_b3 = (const float*)d_in[58];
  const float* fine_w1 = (const float*)d_in[59];  // (1027,1024)
  const float* fine_w2 = (const float*)d_in[60];  // (1024,1024)
  const float* fine_w3 = (const float*)d_in[61];  // (1024,1)
  const float* pw1 = (const float*)d_in[62];  // (64,128)
  const float* pb1 = (const float*)d_in[63];
  const float* pw2 = (const float*)d_in[64];  // (128,128)
  const float* pb2 = (const float*)d_in[65];
  const float* pw3 = (const float*)d_in[66];  // (128,6)
  const float* pb3 = (const float*)d_in[67];
  const float* point_coord = (const float*)d_in[68];
  (void)in_sizes; (void)n_in; (void)out_size; (void)ws_size;

  // ---- workspace carve ----
  float* ws = (float*)d_ws;
  size_t off = 0;
  auto alloc = [&](size_t n) { float* p = ws + off; off += n; return p; };
  int*   fidx  = (int*)alloc(4096);
  float* xyzB0 = alloc(4 * 1024 * 3);
  float* xyzB1 = alloc(4 * 1024 * 3);
  float* ptsB0 = alloc(4 * 1024 * 64);
  float* ptsB1 = alloc(4 * 1024 * 64);
  float* z     = alloc(BS * ZD);
  float* tmpA  = alloc(BS * HID);
  float* tmpB  = alloc(BS * HID);
  float* ezd   = alloc(BS * ZD);
  float* ezg   = alloc(BS * ZD);
  float* ep    = alloc(T_PARTS * PD);
  float* tmp64 = alloc(BS * PD);
  float* zfbuf = alloc(BS * PD);
  float* poseA = alloc(2 * PD);
  float* poseB = alloc(2 * PD);
  float* pp6   = alloc(T_PARTS * 6);
  float* DEF   = alloc((size_t)ROWS * 3);
  float* H2a   = alloc((size_t)ROWS * 256);
  float* H2b   = alloc((size_t)ROWS * 256);
  float* INP   = alloc((size_t)ROWS * 1027);
  float* H1    = alloc((size_t)ROWS * 1024);
  float* H1b   = alloc((size_t)ROWS * 1024);
  float* PFE   = alloc((size_t)ROWS * 1024);

  float* outGS   = (float*)d_out;            // (ROWS,4)
  float* outMax  = outGS + (size_t)ROWS * 4; // (ROWS,1)
  float* outFine = outMax + ROWS;            // (ROWS,1)

  // ---- encoder: 4 SA layers ----
  const int   Ns[4] = {2048, 1024, 256, 64};
  const int   Ss[4] = {1024, 256, 64, 16};
  const float rad[4] = {0.1f, 0.2f, 0.4f, 0.8f};
  const int   Cin[4] = {3, 64, 128, 256};        // pts channels going in
  const int   Cm[4][3] = {{32, 32, 64}, {64, 64, 128}, {128, 128, 256}, {256, 256, ZD}};
  const float* xyzIn = fps_point;
  const float* ptsIn = fps_point;
  float* xyzOutB[2] = {xyzB0, xyzB1};
  float* ptsOutB[2] = {ptsB0, ptsB1};
  for (int l = 0; l < 4; ++l) {
    k_fps<<<BS, 256, 0, stream>>>(xyzIn, Ns[l], Ss[l], fidx);
    k_sa<<<BS * Ss[l], 256, 0, stream>>>(
        xyzIn, ptsIn, Ns[l], Cin[l], fidx, Ss[l], rad[l],
        saw[l][0], sab[l][0], Cm[l][0],
        saw[l][1], sab[l][1], Cm[l][1],
        saw[l][2], sab[l][2], Cm[l][2],
        xyzOutB[l & 1], ptsOutB[l & 1]);
    xyzIn = xyzOutB[l & 1];
    ptsIn = ptsOutB[l & 1];
  }

  // ---- latent z and embed chains (tiny, linear) ----
  k_z<<<4, 256, 0, stream>>>(ptsIn, conv0_w, conv0_b, z);
  k_dense_small<<<1, 256, 0, stream>>>(z, de1w0, de1b0, tmpA, BS, ZD, HID, 0);
  k_dense_small<<<1, 256, 0, stream>>>(tmpA, de1w1, de1b1, tmpB, BS, HID, HID, 0);
  k_dense_small<<<1, 256, 0, stream>>>(tmpB, de1w2, de1b2, ezd, BS, HID, ZD, 0);
  k_dense_small<<<1, 256, 0, stream>>>(z, ge1w0, ge1b0, tmpA, BS, ZD, HID, 0);
  k_dense_small<<<1, 256, 0, stream>>>(tmpA, ge1w1, ge1b1, tmpB, BS, HID, HID, 0);
  k_dense_small<<<1, 256, 0, stream>>>(tmpB, ge1w2, ge1b2, ezg, BS, HID, ZD, 0);
  k_dense_small<<<1, 256, 0, stream>>>(pcodes, emb2w, emb2b, ep, T_PARTS, PD, PD, 0);
  for (int t = 0; t < T_PARTS; ++t) {   // pose MLP per part (1 row)
    k_dense_small<<<1, 256, 0, stream>>>(ep + t * PD, pw1, pb1, poseA, 1, PD, 2 * PD, 4);
    k_dense_small<<<1, 256, 0, stream>>>(poseA, pw2, pb2, poseB, 1, 2 * PD, 2 * PD, 4);
    k_dense_small<<<1, 256, 0, stream>>>(poseB, pw3, pb3, pp6 + t * 6, 1, 2 * PD, 6, 0);
  }

  // ---- per-part deform + BAE (WMMA GEMMs, fused bias/act) ----
  const int gBig = cdiv((long long)ROWS * 67, 256);
  for (int t = 0; t < T_PARTS; ++t) {
    // zf = (ezg_t + ep_t) @ embed3 + b
    k_add_vec<<<1, 256, 0, stream>>>(ezg + t * PD, ZD, ep + t * PD, tmp64, BS, PD);
    k_dense_small<<<1, 256, 0, stream>>>(tmp64, emb3w, emb3b, zfbuf, BS, PD, PD, 0);
    // deform MLP: 67 -> 256 -> 256 -> 3 (tanh)
    k_build_dinp<<<gBig, 256, 0, stream>>>(point_coord, ezd, pcodes, INP, t);
    gemm(stream, INP, 67, d_w1 + (size_t)t * 67 * 256, d_b1 + t * 256, H2a, 256,
         ROWS, 256, 67, 1);
    gemm(stream, H2a, 256, d_w2 + (size_t)t * 256 * 256, d_b2 + t * 256, H2b, 256,
         ROWS, 256, 256, 1);
    gemm(stream, H2b, 256, d_w3 + (size_t)t * 256 * 3, d_b3 + t * 3, DEF, 3,
         ROWS, 3, 256, 3);
    // BAE MLP: 67 -> 1024 -> 256 (pfe slice) -> 1 (Gs column, sigmoid)
    k_build_binp<<<gBig, 256, 0, stream>>>(point_coord, DEF, pp6 + t * 6, zfbuf, INP);
    gemm(stream, INP, 67, g_bw1 + (size_t)t * 67 * 1024, g_bb1 + t * 1024, H1, 1024,
         ROWS, 1024, 67, 1);
    gemm(stream, H1, 1024, g_bw2 + (size_t)t * 1024 * 256, g_bb2 + t * 256,
         PFE + t * 256, 1024, ROWS, 256, 1024, 1);
    gemm(stream, PFE + t * 256, 1024, g_bw3 + (size_t)t * 256, g_bb3 + t,
         outGS + t, 4, ROWS, 1, 256, 2);
  }

  // ---- coarse max + fine MLP ----
  k_maxgs<<<cdiv(ROWS, 256), 256, 0, stream>>>(outGS, outMax);
  k_build_fine<<<cdiv((long long)ROWS * 1027, 256), 256, 0, stream>>>(point_coord, PFE, INP);
  gemm(stream, INP, 1027, fine_w1, fine_b1, H1, 1024, ROWS, 1024, 1027, 1);
  gemm(stream, H1, 1024, fine_w2, fine_b2, H1b, 1024, ROWS, 1024, 1024, 1);
  gemm(stream, H1b, 1024, fine_w3, fine_b3, outFine, 1, ROWS, 1, 1024, 2);
}
